// ScaledDotProductAttention_9354438771232
// MI455X (gfx1250) — matrix operand order repaired and checked by reading, not compiled, not run on hardware
//
#include <hip/hip_runtime.h>
#include <hip/hip_bf16.h>
#include <math.h>

// ---------------------------------------------------------------------------
// Scaled dot-product attention for MI455X (gfx1250, wave32, WMMA).
//   out   [2,16,2048,64]   = softmax(QK^T/8) V
//   wgt   [2,16,2048,2048] = softmax(QK^T/8)
// Strategy: 2-pass (online max/sum, then recompute + emit weights + PV),
// bf16 WMMA with hi/lo split for near-fp32 QK^T accuracy.
// ---------------------------------------------------------------------------

typedef __bf16 bf16_t;
typedef bf16_t v16bf __attribute__((ext_vector_type(16)));
typedef bf16_t v8bf  __attribute__((ext_vector_type(8)));
typedef float  v8f   __attribute__((ext_vector_type(8)));

#define SEQ    2048
#define DK     64
#define NBH    32          // batch*heads
#define CHUNK  128         // keys staged in LDS per iteration
#define KP     72          // sK row pitch in bf16 (16B aligned, conflict-free)
#define VP     136         // sV row pitch in bf16
#define WP     40          // per-wave weight staging pitch
#define QROWS  128         // query rows per block (8 waves * 16)

__device__ __forceinline__ v8f wmma_bf16(v16bf a, v16bf b, v8f c) {
  // D = A(16x32 bf16) * B(32x16 bf16) + C(16x16 f32)
  return __builtin_amdgcn_wmma_f32_16x16x32_bf16(
      /*neg_a=*/false, a, /*neg_b=*/false, b,
      /*c_mod=*/(short)0, c, /*reuse_a=*/false, /*reuse_b=*/false);
}

__device__ __forceinline__ v16bf ld16(const bf16_t* p0, const bf16_t* p1) {
  v8bf a = *(const v8bf*)p0;
  v8bf b = *(const v8bf*)p1;
  v16bf r;
#pragma unroll
  for (int i = 0; i < 8; ++i) { r[i] = a[i]; r[8 + i] = b[i]; }
  return r;
}

__global__ __launch_bounds__(256)
void sdpa_wmma_kernel(const float* __restrict__ Q, const float* __restrict__ K,
                      const float* __restrict__ V, float* __restrict__ Out,
                      float* __restrict__ Wgt) {
  __shared__ __align__(16) bf16_t sKhi[CHUNK * KP];   // K chunk, hi bf16
  __shared__ __align__(16) bf16_t sKlo[CHUNK * KP];   // K chunk, lo residual
  __shared__ __align__(16) bf16_t sV[DK * VP];        // V chunk, transposed [d][key]
  __shared__ __align__(16) bf16_t sW[8][16 * WP];     // per-wave weight staging 16x32

  const int lane = threadIdx.x & 31;
  const int wave = threadIdx.x >> 5;
  const int bh   = blockIdx.y;                // 0..31
  const int q0   = blockIdx.x * QROWS;        // block's query base
  const int half = lane >> 4;                 // 0/1: lane group
  const int ln   = lane & 15;

  const float scale = 0.125f;                 // 1/sqrt(64)

  const float* Qb = Q + (size_t)bh * SEQ * DK;
  const float* Kb = K + (size_t)bh * SEQ * DK;
  const float* Vb = V + (size_t)bh * SEQ * DK;
  float*       Ob = Out + (size_t)bh * SEQ * DK;
  float*       Wb = Wgt + (size_t)bh * SEQ * SEQ;

  // ---- load this wave's 16 Q rows into WMMA A-layout registers, hi/lo split
  // A operand, k order used here (same as B): lane holds row M=lane%16; elem s -> K=16*(lane/16)+s.
  const int qrow = q0 + wave * 16 + ln;
  v16bf qhi[2], qlo[2];
#pragma unroll
  for (int w = 0; w < 2; ++w) {
#pragma unroll
    for (int part = 0; part < 2; ++part) {
      const int d0 = w * 32 + half * 16 + part * 8;
      const float* p = Qb + (size_t)qrow * DK + d0;
#pragma unroll
      for (int i = 0; i < 8; ++i) {
        float  f = p[i];
        bf16_t h = (bf16_t)f;
        qhi[w][part * 8 + i] = h;
        qlo[w][part * 8 + i] = (bf16_t)(f - (float)h);
      }
    }
  }

  // ---- LDS chunk loaders (fp32 -> bf16 hi/lo on the fly) ----
  auto load_k_chunk = [&](int kc) {
    const float* src = Kb + (size_t)kc * DK;
#pragma unroll 2
    for (int idx = threadIdx.x; idx < CHUNK * DK; idx += 256) {
      const int key = idx >> 6, d = idx & 63;
      float  f = src[idx];
      bf16_t h = (bf16_t)f;
      sKhi[key * KP + d] = h;
      sKlo[key * KP + d] = (bf16_t)(f - (float)h);
    }
  };
  auto load_v_chunk = [&](int kc) {
    const float* src = Vb + (size_t)kc * DK;
#pragma unroll 2
    for (int idx = threadIdx.x; idx < CHUNK * DK; idx += 256) {
      const int key = idx >> 6, d = idx & 63;
      sV[d * VP + key] = (bf16_t)src[idx];
    }
  };

  // ---- one 16x16 score tile: S = Q(16x64) . K_tile^T, hi/lo split (3 wmma/32d)
  // B operand, k order used here (same as A): lane holds col N=lane%16; K = 16*(lane/16) + elem.
  auto score_tile = [&](int j) -> v8f {
    v8f c = {};
#pragma unroll
    for (int w = 0; w < 2; ++w) {
      const int koff = (j * 16 + ln) * KP + half * 16 + w * 32;
      v16bf bh16 = ld16(&sKhi[koff], &sKhi[koff + 8]);
      v16bf bl16 = ld16(&sKlo[koff], &sKlo[koff + 8]);
      c = wmma_bf16(qlo[w], bh16, c);   // small terms first
      c = wmma_bf16(qhi[w], bl16, c);
      c = wmma_bf16(qhi[w], bh16, c);
    }
    return c;
  };

  // ======================= PASS 1: online row max / expsum ==================
  // Per-lane partials only (C layout: elem r belongs to row r + 8*half).
  float pm[8], pl[8];
#pragma unroll
  for (int r = 0; r < 8; ++r) { pm[r] = -INFINITY; pl[r] = 0.f; }

  for (int kc = 0; kc < SEQ; kc += CHUNK) {
    load_k_chunk(kc);
    if (kc + CHUNK < SEQ)  // pull next chunk toward L2 (global_prefetch_b8)
      __builtin_prefetch(Kb + (size_t)(kc + CHUNK) * DK + threadIdx.x * 32, 0, 0);
    __syncthreads();
#pragma unroll
    for (int j = 0; j < CHUNK / 16; ++j) {
      v8f c = score_tile(j);
#pragma unroll
      for (int r = 0; r < 8; ++r) {
        float s  = c[r] * scale;
        float mo = pm[r];
        float mn = fmaxf(mo, s);
        pl[r] = pl[r] * __expf(mo - mn) + __expf(s - mn);
        pm[r] = mn;
      }
    }
    __syncthreads();
  }

  // Cross-lane combine within each 16-lane group -> per-row max & 1/sum.
  float rm[8], rinv[8];
#pragma unroll
  for (int r = 0; r < 8; ++r) {
    float m = pm[r], l = pl[r];
#pragma unroll
    for (int off = 1; off < 16; off <<= 1) {
      float om = __shfl_xor(m, off, 16);
      float ol = __shfl_xor(l, off, 16);
      float mn = fmaxf(m, om);
      l = l * __expf(m - mn) + ol * __expf(om - mn);
      m = mn;
    }
    rm[r]   = m;
    rinv[r] = 1.0f / l;
  }

  // ============ PASS 2: recompute, emit weights, accumulate O = W.V =========
  v8f oacc[4] = {};                       // 4 d-chunks of 16 cols
  bf16_t* myW = sW[wave];

  for (int kc = 0; kc < SEQ; kc += CHUNK) {
    load_k_chunk(kc);
    load_v_chunk(kc);
    if (kc + CHUNK < SEQ) {
      __builtin_prefetch(Kb + (size_t)(kc + CHUNK) * DK + threadIdx.x * 32, 0, 0);
      __builtin_prefetch(Vb + (size_t)(kc + CHUNK) * DK + threadIdx.x * 32, 0, 0);
    }
    __syncthreads();
#pragma unroll
    for (int j2 = 0; j2 < CHUNK / 32; ++j2) {
#pragma unroll
      for (int jt = 0; jt < 2; ++jt) {
        const int j = j2 * 2 + jt;
        v8f c = score_tile(j);
#pragma unroll
        for (int r = 0; r < 8; ++r) {
          float s   = c[r] * scale;
          float wgt = __expf(s - rm[r]) * rinv[r];
          const int m = r + half * 8;   // C layout row
          Wb[(size_t)(q0 + wave * 16 + m) * SEQ + (kc + j * 16 + ln)] = wgt;
          myW[m * WP + jt * 16 + ln] = (bf16_t)wgt;  // bf16 stage for PV A-operand
        }
      }
      // PV: A = staged weights (16 rows x 32 keys), B = sV (32 keys x 16 d).
      // Same-wave DS ops are in-order, so no barrier needed for myW reuse.
      const int wb = ln * WP + half * 16;
      v16bf a = ld16(&myW[wb], &myW[wb + 8]);
#pragma unroll
      for (int n4 = 0; n4 < 4; ++n4) {
        const int voff = (n4 * 16 + ln) * VP + j2 * 32 + half * 16;
        v16bf b = ld16(&sV[voff], &sV[voff + 8]);
        oacc[n4] = wmma_bf16(a, b, oacc[n4]);
      }
    }
    __syncthreads();
  }

  // ---- store O (C layout: row r + 8*half, col lane%16 per 16-col chunk) ----
#pragma unroll
  for (int n4 = 0; n4 < 4; ++n4) {
#pragma unroll
    for (int r = 0; r < 8; ++r) {
      const int m = r + half * 8;
      Ob[(size_t)(q0 + wave * 16 + m) * DK + (n4 * 16 + ln)] = oacc[n4][r];
    }
  }
}

extern "C" void kernel_launch(void* const* d_in, const int* in_sizes, int n_in,
                              void* d_out, int out_size, void* d_ws, size_t ws_size,
                              hipStream_t stream) {
  (void)in_sizes; (void)n_in; (void)out_size; (void)d_ws; (void)ws_size;
  const float* Q = (const float*)d_in[0];
  const float* K = (const float*)d_in[1];
  const float* V = (const float*)d_in[2];
  float* Out = (float*)d_out;
  float* Wgt = Out + (size_t)NBH * SEQ * DK;     // outputs concatenated: (output, weights)

  dim3 grid(SEQ / QROWS, NBH);                    // (16, 32)
  sdpa_wmma_kernel<<<grid, 256, 0, stream>>>(Q, K, V, Out, Wgt);
}
